// MultiHeadAttention_15058155340232
// MI455X (gfx1250) — matrix operand order repaired and checked by reading, not compiled, not run on hardware
//
#include <hip/hip_runtime.h>
#include <hip/hip_bf16.h>
#include <math.h>
#include <stdint.h>

// ---------------------------------------------------------------------------
// MHA forward for MI455X (gfx1250, wave32).
// ~103 GFLOP vs ~50MB traffic @ 23.3 TB/s -> strongly compute bound.
// bf16 operands + v_wmma_f32_16x16x32_bf16 (f32 accumulate) everywhere;
// fused causal flash-attention with block-cooperative K/V staging through LDS
// via global_load_async_to_lds_b128 (ASYNCcnt) for 4x K/V reuse.
// ---------------------------------------------------------------------------

typedef __bf16 bf16_t;
typedef __attribute__((ext_vector_type(16))) __bf16 v16bf;
typedef __attribute__((ext_vector_type(8)))  float  v8f;

#define B_    2
#define S_    2048
#define E_    1024
#define H_    16
#define D_    64
#define MTOK  (B_ * S_)       // 4096 tokens
#define NQKV  (3 * E_)        // 3072

// -------------------- WMMA helpers (layouts per CDNA5 ISA 7.12.2) ----------

__device__ __forceinline__ v8f wmma_bf16(v16bf a, v16bf b, v8f c) {
  return __builtin_amdgcn_wmma_f32_16x16x32_bf16(false, a, false, b,
                                                 (short)0, c, false, false);
}

// A-matrix 16x32 bf16 fragment from row-major source (leading dim lda).
__device__ __forceinline__ v16bf load_a_frag(const bf16_t* __restrict__ A,
                                             int lda, int row0, int k0,
                                             int lane) {
  const int row  = row0 + (lane & 15);
  const int half = lane >> 4;
  const bf16_t* p0 = A + (size_t)row * lda + k0 + half * 8;
  const bf16_t* p1 = A + (size_t)row * lda + k0 + 16 + half * 8;
  v16bf a;
#pragma unroll
  for (int e = 0; e < 8; ++e) a[e] = p0[e];       // K = half*8 + e
#pragma unroll
  for (int e = 0; e < 8; ++e) a[8 + e] = p1[e];   // K = 16 + half*8 + e
  return a;
}

// B-matrix 32x16 (KxN) fragment from a TRANSPOSED source BT (N x K row-major):
// each lane reads two runs of 8 contiguous bf16 (16B each). Works for global or LDS pointers.
__device__ __forceinline__ v16bf load_bT_frag(const bf16_t* __restrict__ BT,
                                              int ldk, int k0, int col0,
                                              int lane) {
  const int col  = col0 + (lane & 15);
  const int half = lane >> 4;
  const bf16_t* p = BT + (size_t)col * ldk + k0 + half * 8;
  v16bf b;
#pragma unroll
  for (int e = 0; e < 16; ++e) b[e] = p[e < 8 ? e : e + 8];  // K = half*8 + e (e<8), 16 + half*8 + (e-8) (e>=8)
  return b;
}

__device__ __forceinline__ float rowmax16(float v) {
#pragma unroll
  for (int off = 1; off < 16; off <<= 1) v = fmaxf(v, __shfl_xor(v, off, 16));
  return v;
}
__device__ __forceinline__ float rowsum16(float v) {
#pragma unroll
  for (int off = 1; off < 16; off <<= 1) v += __shfl_xor(v, off, 16);
  return v;
}

// Async global->LDS copy (16B per lane), tracked by ASYNCcnt.
__device__ __forceinline__ void async_load_b128(uint32_t lds_off,
                                                const void* gptr) {
  asm volatile("global_load_async_to_lds_b128 %0, %1, off"
               :: "v"(lds_off), "v"((unsigned long long)(uintptr_t)gptr)
               : "memory");
}
__device__ __forceinline__ void wait_async0() {
  asm volatile("s_wait_asynccnt 0" ::: "memory");
}

// -------------------- Stage 0: fp32 -> bf16 converts -----------------------

__global__ void cvt_kernel(const float* __restrict__ in,
                           bf16_t* __restrict__ out, int n) {
  int i = blockIdx.x * blockDim.x + threadIdx.x;
  if (i < n) out[i] = (bf16_t)in[i];
}

__global__ void cvtT_kernel(const float* __restrict__ in,
                            bf16_t* __restrict__ out, int rows, int cols) {
  int i = blockIdx.x * blockDim.x + threadIdx.x;
  if (i < rows * cols) {
    int r = i / cols, c = i % cols;
    out[(size_t)c * rows + r] = (bf16_t)in[i];
  }
}

// -------------------- Stage 1: QKV projection ------------------------------
// Wave tile 32x64: 2 A-frags x 4 B-frags -> 8 WMMAs per 32-deep k-step.
__global__ __launch_bounds__(128)
void gemm_qkv_kernel(const bf16_t* __restrict__ A,
                     const bf16_t* __restrict__ BT,
                     bf16_t* __restrict__ Qo, bf16_t* __restrict__ Ko,
                     bf16_t* __restrict__ Vo) {
  const int lane = threadIdx.x & 31;
  const int widx = threadIdx.x >> 5;
  const int row0 = blockIdx.y * 128 + widx * 32;
  const int col0 = blockIdx.x * 64;

  v8f acc[2][4] = {};
#pragma unroll 2
  for (int k0 = 0; k0 < E_; k0 += 32) {
    __builtin_prefetch(A + (size_t)row0 * E_ + k0 + 128, 0, 1);
    v16bf a0 = load_a_frag(A, E_, row0, k0, lane);
    v16bf a1 = load_a_frag(A, E_, row0 + 16, k0, lane);
#pragma unroll
    for (int t = 0; t < 4; ++t) {
      v16bf b = load_bT_frag(BT, E_, k0, col0 + t * 16, lane);
      acc[0][t] = wmma_bf16(a0, b, acc[0][t]);
      acc[1][t] = wmma_bf16(a1, b, acc[1][t]);
    }
  }

  const int half = lane >> 4;
#pragma unroll
  for (int g = 0; g < 2; ++g) {
#pragma unroll
    for (int t = 0; t < 4; ++t) {
#pragma unroll
      for (int r = 0; r < 8; ++r) {
        const int tok = row0 + g * 16 + r + half * 8;
        const int col = col0 + t * 16 + (lane & 15);
        const int bi = tok / S_, s = tok % S_;
        const int part = col / E_;
        const int h = (col % E_) / D_;
        const int d = col % D_;
        const size_t bh = (size_t)(bi * H_ + h);
        const bf16_t val = (bf16_t)acc[g][t][r];
        if (part == 0)      Qo[(bh * S_ + s) * D_ + d] = val;
        else if (part == 1) Ko[(bh * S_ + s) * D_ + d] = val;
        else                Vo[(bh * D_ + d) * S_ + s] = val;  // V transposed
      }
    }
  }
}

// -------------------- Stage 2: fused causal flash attention ----------------
// Block = 4 waves sharing one (b,h); wave w handles q-block qsuper*4+w.
// Each 32-key iteration: all threads async-stage K tile (32x64) and V tile
// (64x32, depth-major) into LDS; every wave consumes them (4x reuse).
__global__ __launch_bounds__(128)
void attn_kernel(const bf16_t* __restrict__ Q, const bf16_t* __restrict__ K,
                 const bf16_t* __restrict__ V, bf16_t* __restrict__ O) {
  __shared__ __align__(16) bf16_t kTile[32 * D_];     // 4KB (key-major)
  __shared__ __align__(16) bf16_t vTile[D_ * 32];     // 4KB (depth-major)
  __shared__ __align__(16) bf16_t pst[4 * 16 * 32];   // 4KB P re-layout

  const int tid  = threadIdx.x;
  const int lane = tid & 31;
  const int widx = tid >> 5;
  const int qsuper = blockIdx.x & 31;     // S_/64 = 32 q-supertiles
  const int bh     = blockIdx.x >> 5;     // 0..B_*H_-1
  const int h = bh & (H_ - 1);
  const int b = bh >> 4;

  const bf16_t* Qbh = Q + (size_t)bh * S_ * D_;   // (s,d)
  const bf16_t* Kbh = K + (size_t)bh * S_ * D_;   // (s,d)
  const bf16_t* Vbh = V + (size_t)bh * D_ * S_;   // (d,s)
  bf16_t* pl = pst + widx * (16 * 32);

  const int half = lane >> 4;
  const int ncol = lane & 15;
  const int qb = qsuper * 4 + widx;
  const int qrow_base = qb * 16 + half * 8;

  v16bf qa0 = load_a_frag(Qbh, D_, qb * 16, 0, lane);
  v16bf qa1 = load_a_frag(Qbh, D_, qb * 16, 32, lane);

  v8f o0 = {}, o1 = {}, o2 = {}, o3 = {};
  float m[8], l[8];
#pragma unroll
  for (int r = 0; r < 8; ++r) { m[r] = -__builtin_inff(); l[r] = 0.f; }

  const int nkb_w   = (qb >> 1) + 1;      // this wave's causal trip count
  const int nkb_max = qsuper * 2 + 2;     // block max (last wave's count)

  for (int kb = 0; kb < nkb_max; ++kb) {
    const int key0 = kb * 32;
    // ---- cooperative async staging: 256 x 16B chunks per tile ----
    {
      const char* kg = (const char*)(Kbh + (size_t)key0 * D_);
      const char* vg = (const char*)Vbh + (size_t)key0 * 2;
#pragma unroll
      for (int i = 0; i < 2; ++i) {
        const int c = tid + i * 128;      // 0..255
        async_load_b128((uint32_t)(uintptr_t)((char*)kTile + c * 16),
                        kg + (size_t)c * 16);
        const int d = c >> 2, cc = c & 3;
        async_load_b128((uint32_t)(uintptr_t)((char*)vTile + d * 64 + cc * 16),
                        vg + (size_t)d * S_ * 2 + cc * 16);
      }
      wait_async0();
      __syncthreads();
    }

    if (kb < nkb_w) {                     // wave-uniform: EXEC stays full
      // ---- S = Q K^T from LDS K tile ----
      v8f s0 = {}, s1 = {};
      s0 = wmma_bf16(qa0, load_bT_frag(kTile, D_, 0,  0,  lane), s0);
      s0 = wmma_bf16(qa1, load_bT_frag(kTile, D_, 32, 0,  lane), s0);
      s1 = wmma_bf16(qa0, load_bT_frag(kTile, D_, 0,  16, lane), s1);
      s1 = wmma_bf16(qa1, load_bT_frag(kTile, D_, 32, 16, lane), s1);

      // ---- online softmax ----
#pragma unroll
      for (int r = 0; r < 8; ++r) {
        const int q = qrow_base + r;
        float x0 = s0[r] * 0.125f;        // 1/sqrt(64)
        float x1 = s1[r] * 0.125f;
        if (key0 + ncol > q)      x0 = -1e9f;
        if (key0 + 16 + ncol > q) x1 = -1e9f;
        const float mx = rowmax16(fmaxf(x0, x1));
        const float mn = fmaxf(m[r], mx);
        const float corr = __expf(m[r] - mn);
        const float p0 = __expf(x0 - mn);
        const float p1 = __expf(x1 - mn);
        l[r] = l[r] * corr + rowsum16(p0 + p1);
        m[r] = mn;
        o0[r] *= corr; o1[r] *= corr; o2[r] *= corr; o3[r] *= corr;
        pl[(r + half * 8) * 32 + ncol]      = (bf16_t)p0;
        pl[(r + half * 8) * 32 + 16 + ncol] = (bf16_t)p1;
      }
      // ---- P (C-layout) -> A-fragment via wave-private LDS ----
      v16bf pa;
      {
        const bf16_t* ap0 = pl + (lane & 15) * 32 + half * 8;
        const bf16_t* ap1 = pl + (lane & 15) * 32 + 16 + half * 8;
#pragma unroll
        for (int e = 0; e < 8; ++e) pa[e] = ap0[e];
#pragma unroll
        for (int e = 0; e < 8; ++e) pa[8 + e] = ap1[e];
      }
      // ---- O += P V from LDS V tile (depth-major, ld=32) ----
      o0 = wmma_bf16(pa, load_bT_frag(vTile, 32, 0, 0,  lane), o0);
      o1 = wmma_bf16(pa, load_bT_frag(vTile, 32, 0, 16, lane), o1);
      o2 = wmma_bf16(pa, load_bT_frag(vTile, 32, 0, 32, lane), o2);
      o3 = wmma_bf16(pa, load_bT_frag(vTile, 32, 0, 48, lane), o3);
    }
    __syncthreads();                      // protect tiles before next stage
  }

  // ---- epilogue: normalize, store (b, s, e) bf16 for out-proj ----
#pragma unroll
  for (int r = 0; r < 8; ++r) {
    const int q = qrow_base + r;
    const size_t tok = (size_t)(b * S_ + q);
    const float inv = 1.0f / l[r];
    O[tok * E_ + h * D_ + 0  + ncol] = (bf16_t)(o0[r] * inv);
    O[tok * E_ + h * D_ + 16 + ncol] = (bf16_t)(o1[r] * inv);
    O[tok * E_ + h * D_ + 32 + ncol] = (bf16_t)(o2[r] * inv);
    O[tok * E_ + h * D_ + 48 + ncol] = (bf16_t)(o3[r] * inv);
  }
}

// -------------------- Stage 3: output projection ---------------------------
__global__ __launch_bounds__(128)
void gemm_proj_kernel(const bf16_t* __restrict__ A,
                      const bf16_t* __restrict__ BT,
                      float* __restrict__ C) {
  const int lane = threadIdx.x & 31;
  const int widx = threadIdx.x >> 5;
  const int row0 = blockIdx.y * 128 + widx * 32;
  const int col0 = blockIdx.x * 64;

  v8f acc[2][4] = {};
#pragma unroll 2
  for (int k0 = 0; k0 < E_; k0 += 32) {
    __builtin_prefetch(A + (size_t)row0 * E_ + k0 + 128, 0, 1);
    v16bf a0 = load_a_frag(A, E_, row0, k0, lane);
    v16bf a1 = load_a_frag(A, E_, row0 + 16, k0, lane);
#pragma unroll
    for (int t = 0; t < 4; ++t) {
      v16bf b = load_bT_frag(BT, E_, k0, col0 + t * 16, lane);
      acc[0][t] = wmma_bf16(a0, b, acc[0][t]);
      acc[1][t] = wmma_bf16(a1, b, acc[1][t]);
    }
  }

  const int half = lane >> 4;
#pragma unroll
  for (int g = 0; g < 2; ++g)
#pragma unroll
    for (int t = 0; t < 4; ++t)
#pragma unroll
      for (int r = 0; r < 8; ++r) {
        const int row = row0 + g * 16 + r + half * 8;
        const int col = col0 + t * 16 + (lane & 15);
        C[(size_t)row * E_ + col] = acc[g][t][r];
      }
}

// -------------------- host-side launch -------------------------------------

extern "C" void kernel_launch(void* const* d_in, const int* in_sizes, int n_in,
                              void* d_out, int out_size, void* d_ws,
                              size_t ws_size, hipStream_t stream) {
  const float* x     = (const float*)d_in[0];   // (B,S,E)
  const float* Wqkv  = (const float*)d_in[1];   // (E, 3E)
  const float* Wproj = (const float*)d_in[2];   // (E, E)
  // d_in[3] = causal mask; handled analytically in attn_kernel.

  char* ws = (char*)d_ws;
  auto take = [&](size_t bytes) -> char* {
    char* p = ws;
    ws += (bytes + 255) & ~(size_t)255;
    return p;
  };
  bf16_t* xb     = (bf16_t*)take((size_t)MTOK * E_ * 2);
  bf16_t* WqkvT  = (bf16_t*)take((size_t)NQKV * E_ * 2);
  bf16_t* WprojT = (bf16_t*)take((size_t)E_ * E_ * 2);
  bf16_t* Qm     = (bf16_t*)take((size_t)B_ * H_ * S_ * D_ * 2);
  bf16_t* Km     = (bf16_t*)take((size_t)B_ * H_ * S_ * D_ * 2);
  bf16_t* Vm     = (bf16_t*)take((size_t)B_ * H_ * S_ * D_ * 2);
  bf16_t* attn   = (bf16_t*)take((size_t)MTOK * E_ * 2);

  {
    int n = MTOK * E_;
    cvt_kernel<<<(n + 255) / 256, 256, 0, stream>>>(x, xb, n);
  }
  {
    int n = E_ * NQKV;
    cvtT_kernel<<<(n + 255) / 256, 256, 0, stream>>>(Wqkv, WqkvT, E_, NQKV);
  }
  {
    int n = E_ * E_;
    cvtT_kernel<<<(n + 255) / 256, 256, 0, stream>>>(Wproj, WprojT, E_, E_);
  }

  gemm_qkv_kernel<<<dim3(NQKV / 64, MTOK / 128), 128, 0, stream>>>(
      xb, WqkvT, Qm, Km, Vm);

  attn_kernel<<<B_ * H_ * (S_ / 64), 128, 0, stream>>>(Qm, Km, Vm, attn);

  gemm_proj_kernel<<<dim3(E_ / 64, MTOK / 128), 128, 0, stream>>>(
      attn, WprojT, (float*)d_out);
}
